// MaskedMultiHeadAttention_11519102287883
// MI455X (gfx1250) — matrix operand order repaired and checked by reading, not compiled, not run on hardware
//
#include <hip/hip_runtime.h>
#include <hip/hip_bf16.h>

typedef __bf16 bf16;
typedef __attribute__((ext_vector_type(16))) __bf16 v16bf;
typedef __attribute__((ext_vector_type(8)))  __bf16 v8bf;
typedef __attribute__((ext_vector_type(8)))  float  v8f;

union FragU { v16bf v; v8bf h[2]; };

#define WMMA_BF16(a,b,c) \
  __builtin_amdgcn_wmma_f32_16x16x32_bf16(false,(a),false,(b),(short)0,(c),false,false)

// A fragment (16x32, bf16): lane L holds row m = L&15.
// lanes 0-15: elems 0..7 = K[0..7],  elems 8..15 = K[16..23]
// lanes16-31: elems 0..7 = K[8..15], elems 8..15 = K[24..31]
__device__ __forceinline__ v16bf load_a_frag(const bf16* __restrict__ base, int ld,
                                             int m, int k0, int lane) {
  const int row = m + (lane & 15);
  const int kb  = k0 + ((lane & 16) ? 8 : 0);
  FragU f;
  f.h[0] = *(const v8bf*)(base + (size_t)row * ld + kb);
  f.h[1] = *(const v8bf*)(base + (size_t)row * ld + kb + 16);
  return f.v;
}

// B fragment (32x16, bf16): lane L holds column n = L&15; same K order as A fragment.
// B(k,n) = base[n*ld + k].
__device__ __forceinline__ v16bf load_b_frag(const bf16* __restrict__ base, int ld,
                                             int n, int k0, int lane) {
  const int col = n + (lane & 15);
  const int kb  = k0 + ((lane & 16) ? 8 : 0);
  FragU f;
  f.h[0] = *(const v8bf*)(base + (size_t)col * ld + kb);
  f.h[1] = *(const v8bf*)(base + (size_t)col * ld + kb + 16);
  return f.v;
}

// ---------------------------------------------------------------------------
// GEMM: out[m,n] = sum_k A[m,k] * W[n,k] + bias[n]   (y = A @ W^T)
// Wave tile 64x64 (4 M-tiles x 4 N-tiles): 16 WMMA per 16 b128 loads per K-step.
// Block = 4 waves in 2x2 -> 128x128 block tile.
// ---------------------------------------------------------------------------
template <bool OUT_BF16>
__global__ void __launch_bounds__(128)
gemm_bf16_wmma(const bf16* __restrict__ A, const bf16* __restrict__ W,
               const float* __restrict__ bias, void* __restrict__ out,
               int M, int N, int K) {
  const int lane = threadIdx.x & 31;
  const int widx = threadIdx.x >> 5;
  const int m0 = (blockIdx.x * 2 + (widx & 1)) * 64;
  const int n0 = (blockIdx.y * 2 + (widx >> 1)) * 64;

  // Per-lane base pointers (A-layout / B-layout gathers).
  const bf16* ap = A + (size_t)(m0 + (lane & 15)) * K + ((lane & 16) ? 8 : 0);
  const bf16* wp = W + (size_t)(n0 + (lane & 15)) * K + ((lane & 16) ? 8 : 0);

  v8f acc[4][4] = {};
  for (int k0 = 0; k0 < K; k0 += 32) {
    v16bf a[4], b[4];
#pragma unroll
    for (int i = 0; i < 4; ++i) {
      FragU f;
      f.h[0] = *(const v8bf*)(ap + (size_t)(16 * i) * K + k0);
      f.h[1] = *(const v8bf*)(ap + (size_t)(16 * i) * K + k0 + 16);
      a[i] = f.v;
    }
#pragma unroll
    for (int j = 0; j < 4; ++j) {
      FragU f;
      f.h[0] = *(const v8bf*)(wp + (size_t)(16 * j) * K + k0);
      f.h[1] = *(const v8bf*)(wp + (size_t)(16 * j) * K + k0 + 16);
      b[j] = f.v;
    }
    if (k0 + 64 < K) {  // hint next K-tile into cache (global_prefetch_b8)
      __builtin_prefetch(ap + k0 + 64, 0, 1);
      __builtin_prefetch(wp + k0 + 64, 0, 1);
    }
#pragma unroll
    for (int i = 0; i < 4; ++i)
#pragma unroll
      for (int j = 0; j < 4; ++j)
        acc[i][j] = WMMA_BF16(a[i], b[j], acc[i][j]);
  }

  // C/D layout: VGPR r -> row r (lanes 0-15) / row r+8 (lanes 16-31); lane&15 = col.
  const int r8 = (lane & 16) ? 8 : 0;
  const int cb = lane & 15;
#pragma unroll
  for (int j = 0; j < 4; ++j) {
    const float bz = bias[n0 + 16 * j + cb];
#pragma unroll
    for (int i = 0; i < 4; ++i) {
#pragma unroll
      for (int r = 0; r < 8; ++r) {
        const int row = m0 + 16 * i + r + r8;
        const int col = n0 + 16 * j + cb;
        const float v = acc[i][j][r] + bz;
        if (OUT_BF16) ((bf16*)out)[(size_t)row * N + col] = (bf16)v;
        else          ((float*)out)[(size_t)row * N + col] = v;
      }
    }
  }
}

// ---------------------------------------------------------------------------
// Flash attention, causal. One wave per (b, h, 16-query tile), key-step 64.
// Q,K: bf16 [B,H,T,64]; Vt: bf16 [B,H,64,T]; Out: bf16 [B*T, C].
// ---------------------------------------------------------------------------
__global__ void __launch_bounds__(128)
attn_flash_wmma(const bf16* __restrict__ Q, const bf16* __restrict__ Kc,
                const bf16* __restrict__ Vt, bf16* __restrict__ Out,
                int B, int H, int T) {
  const int lane = threadIdx.x & 31;
  const int widx = threadIdx.x >> 5;
  const int w    = blockIdx.x * 4 + widx;
  const int nq   = T / 16;
  const int qt   = w % nq;
  const int bh   = w / nq;
  const int b    = bh / H;
  const int h    = bh % H;
  const int q0   = qt * 16;

  const bf16* qp = Q  + (size_t)bh * T * 64;
  const bf16* kp = Kc + (size_t)bh * T * 64;
  const bf16* vp = Vt + (size_t)bh * 64 * T;

  __shared__ __align__(16) bf16 pbuf[4][16][64];  // per-wave P repack buffer (8 KB)

  v16bf qa[2];
  qa[0] = load_a_frag(qp, 64, q0, 0,  lane);
  qa[1] = load_a_frag(qp, 64, q0, 32, lane);

  v8f o[4] = {};
  float mrow[8], lrow[8];
#pragma unroll
  for (int r = 0; r < 8; ++r) { mrow[r] = -1e30f; lrow[r] = 0.f; }

  const int r8 = (lane & 16) ? 8 : 0;
  const int cb = lane & 15;
  const float scl = 0.03125f;  // C^-0.5 = 1/sqrt(1024)
  const int qlast = q0 + 15;   // last valid row of this tile

  for (int jb = 0; jb < q0 + 16; jb += 64) {
    // ---- S[16 x 64] = Q (16x64) @ K^T (64x64), four 16x16 C tiles ----
    v8f s[4];
#pragma unroll
    for (int t = 0; t < 4; ++t) {
      const int kt0 = jb + 16 * t;
      if (kt0 > qlast) {               // tile fully above the diagonal -> P = 0
#pragma unroll
        for (int r = 0; r < 8; ++r) s[t][r] = -1e30f;
        continue;
      }
      v8f z = {};
      v16bf bk0 = load_b_frag(kp, 64, kt0, 0,  lane);  // B(k=dim, n=key)
      v16bf bk1 = load_b_frag(kp, 64, kt0, 32, lane);
      z = WMMA_BF16(qa[0], bk0, z);
      z = WMMA_BF16(qa[1], bk1, z);
      const bool partial = (kt0 + 15 > q0);            // tile straddles the diagonal
#pragma unroll
      for (int r = 0; r < 8; ++r) {
        float v = z[r] * scl;
        if (partial) {
          const int key = kt0 + cb;
          const int row = q0 + r + r8;
          if (key > row) v = -1e30f;
        }
        s[t][r] = v;
      }
    }

    // ---- online softmax (rows live across lanes within 16-lane half-groups) ----
#pragma unroll
    for (int r = 0; r < 8; ++r) {
      float mx = fmaxf(fmaxf(s[0][r], s[1][r]), fmaxf(s[2][r], s[3][r]));
#pragma unroll
      for (int d = 1; d < 16; d <<= 1) mx = fmaxf(mx, __shfl_xor(mx, d, 32));
      const float mnew  = fmaxf(mrow[r], mx);
      const float alpha = __expf(mrow[r] - mnew);
      float rs = 0.f;
#pragma unroll
      for (int t = 0; t < 4; ++t) {
        const float p = __expf(s[t][r] - mnew);
        s[t][r] = p;
        rs += p;
      }
#pragma unroll
      for (int d = 1; d < 16; d <<= 1) rs += __shfl_xor(rs, d, 32);
      lrow[r] = lrow[r] * alpha + rs;
      mrow[r] = mnew;
#pragma unroll
      for (int j = 0; j < 4; ++j) o[j][r] *= alpha;
    }

    // ---- repack P (16x64): C-layout -> A-layout via per-wave LDS bounce ----
#pragma unroll
    for (int t = 0; t < 4; ++t)
#pragma unroll
      for (int r = 0; r < 8; ++r)
        pbuf[widx][r + r8][16 * t + cb] = (bf16)s[t][r];
    asm volatile("s_wait_dscnt 0" ::: "memory");  // LDS in-order per wave; cross-lane visibility

    FragU pf0, pf1;
    {
      const int prow = lane & 15;
      const int pk   = (lane & 16) ? 8 : 0;
      pf0.h[0] = *(const v8bf*)&pbuf[widx][prow][pk];
      pf0.h[1] = *(const v8bf*)&pbuf[widx][prow][pk + 16];
      pf1.h[0] = *(const v8bf*)&pbuf[widx][prow][32 + pk];
      pf1.h[1] = *(const v8bf*)&pbuf[widx][prow][32 + pk + 16];
    }

    // ---- O (16x64) += P (16x64) @ V (64x64) ----
#pragma unroll
    for (int j = 0; j < 4; ++j) {
      v16bf vb0 = load_b_frag(vp, T, 16 * j, jb,      lane);  // B(k=key, n=dim) from Vt
      v16bf vb1 = load_b_frag(vp, T, 16 * j, jb + 32, lane);
      o[j] = WMMA_BF16(pf0.v, vb0, o[j]);
      o[j] = WMMA_BF16(pf1.v, vb1, o[j]);
    }
  }

  // ---- normalize + store to [B*T, C] (C = H*64) ----
  const int C = H * 64;
#pragma unroll
  for (int j = 0; j < 4; ++j)
#pragma unroll
    for (int r = 0; r < 8; ++r) {
      const float v   = o[j][r] / lrow[r];
      const int  trow = q0 + r + r8;
      Out[((size_t)b * T + trow) * C + h * 64 + 16 * j + cb] = (bf16)v;
    }
}

// ---------------------------------------------------------------------------
__global__ void cvt_f32_bf16(const float* __restrict__ in, bf16* __restrict__ out, long n) {
  long i = (long)blockIdx.x * blockDim.x + threadIdx.x;
  const long stride = (long)gridDim.x * blockDim.x;
  for (; i < n; i += stride) out[i] = (bf16)in[i];
}

// qkv [B*T, 3C] -> Q[B,H,T,64], K[B,H,T,64], Vt[B,H,64,T]
__global__ void repack_qkv(const bf16* __restrict__ qkv, bf16* __restrict__ q,
                           bf16* __restrict__ k, bf16* __restrict__ vt,
                           int B, int T, int H) {
  const int C = H * 64;
  const long n = (long)B * T * 3 * C;
  long i = (long)blockIdx.x * blockDim.x + threadIdx.x;
  const long stride = (long)gridDim.x * blockDim.x;
  for (; i < n; i += stride) {
    const int  col   = (int)(i % (3 * C));
    const long m     = i / (3 * C);
    const int  b     = (int)(m / T), t = (int)(m % T);
    const int  third = col / C, c = col % C, h = c / 64, d = c % 64;
    const bf16 val   = qkv[i];
    const size_t bh  = (size_t)(b * H + h);
    if      (third == 0) q [(bh * T + t) * 64 + d] = val;
    else if (third == 1) k [(bh * T + t) * 64 + d] = val;
    else                 vt[(bh * 64 + d) * T + t] = val;
  }
}

// ---------------------------------------------------------------------------
extern "C" void kernel_launch(void* const* d_in, const int* in_sizes, int n_in,
                              void* d_out, int out_size, void* d_ws, size_t ws_size,
                              hipStream_t stream) {
  const int B = 4, T = 2048, C = 1024, H = 16;
  const float* x      = (const float*)d_in[0];
  const float* w_qkv  = (const float*)d_in[1];
  const float* b_qkv  = (const float*)d_in[2];
  const float* w_proj = (const float*)d_in[3];
  const float* b_proj = (const float*)d_in[4];

  char* ws = (char*)d_ws;
  const size_t MB = 1ull << 20;
  bf16* xb     = (bf16*)(ws);            // 16 MB  x in bf16
  bf16* wqkvb  = (bf16*)(ws + 16 * MB);  //  6 MB
  bf16* wprojb = (bf16*)(ws + 22 * MB);  //  2 MB
  bf16* qkvb   = (bf16*)(ws + 24 * MB);  // 48 MB  qkv (bf16)
  bf16* attb   = qkvb;                   // alias: qkv dead after repack
  bf16* qh     = (bf16*)(ws + 72 * MB);  // 16 MB  Q [B,H,T,64]
  bf16* kh     = (bf16*)(ws + 88 * MB);  // 16 MB  K [B,H,T,64]
  bf16* vth    = (bf16*)(ws + 104 * MB); // 16 MB  Vt [B,H,64,T]  (120 MB total)

  cvt_f32_bf16<<<2048, 256, 0, stream>>>(x,      xb,     (long)B * T * C);
  cvt_f32_bf16<<<1024, 256, 0, stream>>>(w_qkv,  wqkvb,  (long)3 * C * C);
  cvt_f32_bf16<<<512,  256, 0, stream>>>(w_proj, wprojb, (long)C * C);

  // qkv = x @ w_qkv^T + b_qkv  (M=8192, N=3072, K=1024), block tile 128x128
  gemm_bf16_wmma<true><<<dim3(B * T / 128, 3 * C / 128), 128, 0, stream>>>(
      xb, wqkvb, b_qkv, qkvb, B * T, 3 * C, C);

  repack_qkv<<<4096, 256, 0, stream>>>(qkvb, qh, kh, vth, B, T, H);

  // attention: B*H*(T/16) = 8192 waves, 4 per block
  attn_flash_wmma<<<(B * H * (T / 16)) / 4, 128, 0, stream>>>(qh, kh, vth, attb, B, H, T);

  // out = att @ w_proj^T + b_proj  (M=8192, N=1024, K=1024), f32 output
  gemm_bf16_wmma<false><<<dim3(B * T / 128, C / 128), 128, 0, stream>>>(
      attb, wprojb, b_proj, (float*)d_out, B * T, C, C);
}
